// DanceDTW_23983097381100
// MI455X (gfx1250) — compile-verified, matrix operand order checked by reading, not run on hardware
//
#include <hip/hip_runtime.h>
#include <math.h>

typedef __attribute__((ext_vector_type(16))) _Float16 v16h;
typedef __attribute__((ext_vector_type(8)))  _Float16 v8h;
typedef __attribute__((ext_vector_type(4)))  _Float16 v4h;
typedef __attribute__((ext_vector_type(8)))  float    v8f;

#define BATCH 32
#define L 1024
#define F 128
#define BIGF 1000000000.0f

// ---------------------------------------------------------------------------
// Kernel 1: prep. One wave per row of length 128:
//  - each lane loads a float4 (coalesced b128),
//  - converts to f16 and stores a v4h (8B, coalesced) into the f16 copy,
//  - accumulates squares and shfl-reduces the fp32 row norm.
// The f16 copies (8 MB each) stay L2-resident and feed the WMMA GEMM, so the
// fp32->f16 conversion happens exactly once per element instead of 32x.
// ---------------------------------------------------------------------------
__global__ void dtw_prep_kernel(const float* __restrict__ seq,
                                float* __restrict__ sq,
                                _Float16* __restrict__ hseq, int nrows) {
    int wave = blockIdx.x * (blockDim.x >> 5) + (threadIdx.x >> 5);
    int lane = threadIdx.x & 31;
    if (wave >= nrows) return;
    const float4* p = (const float4*)(seq + (size_t)wave * F);
    float4 v = p[lane];                       // lane*4 .. lane*4+3 (32*4 = 128)
    v4h hv;
    hv[0] = (_Float16)v.x; hv[1] = (_Float16)v.y;
    hv[2] = (_Float16)v.z; hv[3] = (_Float16)v.w;
    *(v4h*)(hseq + (size_t)wave * F + lane * 4) = hv;
    float s = v.x * v.x + v.y * v.y + v.z * v.z + v.w * v.w;
    #pragma unroll
    for (int off = 16; off; off >>= 1) s += __shfl_xor(s, off, 32);
    if (lane == 0) sq[wave] = s;
}

// Fragment loader for v_wmma_f32_16x16x32_f16 A/B operands.
// Lane layout (16-bit A 16x32, B symmetric as columns): lanes 0-15 hold
// K = base+[0..8) in halves 0-7 and K = base+16+[0..8) in halves 8-15;
// lanes 16-31 the same shifted by 8 (base0 = kk + (lane>>4)*8).
__device__ __forceinline__ v16h dtw_load_frag(const _Float16* p, int base0) {
    v8h lo = *(const v8h*)(p + base0);        // 16B load
    v8h hi = *(const v8h*)(p + base0 + 16);   // 16B load
    v16h r;
    #pragma unroll
    for (int h = 0; h < 8; ++h) { r[h] = lo[h]; r[h + 8] = hi[h]; }
    return r;
}

// ---------------------------------------------------------------------------
// Kernel 2: batched cost matrix via WMMA, 2x2 tiles (32x32 output) per wave.
//   C[b][i][j] = sqrt(max(sq1[i] + sq2[j] - 2*dot(s1[i], s2[j]), 0))
// 16 chained v_wmma_f32_16x16x32_f16 per wave (K=128, 4 accumulators);
// each A fragment is reused for 2 B fragments and vice-versa.
// Grid = 32 batches * 32 i-blocks * 4 j-groups = 4096 blocks of 8 waves.
// ---------------------------------------------------------------------------
__global__ void __launch_bounds__(256)
dtw_cost_gemm_kernel(const _Float16* __restrict__ h1,
                     const _Float16* __restrict__ h2,
                     const float* __restrict__ sq1, const float* __restrict__ sq2,
                     float* __restrict__ C) {
    int id   = blockIdx.x;
    int b    = id >> 7;          // / (32*4)
    int rem  = id & 127;
    int ib   = rem >> 2;         // i-block 0..31 (32 rows each)
    int jg   = rem & 3;          // j-group 0..3
    int w    = threadIdx.x >> 5; // wave 0..7
    int jb   = jg * 8 + w;       // j-block 0..31 (32 cols each)
    int lane = threadIdx.x & 31;
    int lm   = lane & 15;
    int hi   = lane >> 4;

    const _Float16* a0p = h1 + ((size_t)b * L + (size_t)ib * 32 + lm) * F;
    const _Float16* a1p = a0p + 16 * F;
    const _Float16* b0p = h2 + ((size_t)b * L + (size_t)jb * 32 + lm) * F;
    const _Float16* b1p = b0p + 16 * F;

    v8f acc00 = {}, acc01 = {}, acc10 = {}, acc11 = {};
    #pragma unroll
    for (int kk = 0; kk < F; kk += 32) {
        int base0 = kk + hi * 8;
        v16h a0 = dtw_load_frag(a0p, base0);
        v16h a1 = dtw_load_frag(a1p, base0);
        v16h b0 = dtw_load_frag(b0p, base0);
        v16h b1 = dtw_load_frag(b1p, base0);
        acc00 = __builtin_amdgcn_wmma_f32_16x16x32_f16(false, a0, false, b0,
                                                       (short)0, acc00, false, false);
        acc01 = __builtin_amdgcn_wmma_f32_16x16x32_f16(false, a0, false, b1,
                                                       (short)0, acc01, false, false);
        acc10 = __builtin_amdgcn_wmma_f32_16x16x32_f16(false, a1, false, b0,
                                                       (short)0, acc10, false, false);
        acc11 = __builtin_amdgcn_wmma_f32_16x16x32_f16(false, a1, false, b1,
                                                       (short)0, acc11, false, false);
    }

    // Epilogue: C/D layout -> VGPR r holds M = hi*8 + r, N = lane&15.
    float* Cb = C + (size_t)b * L * L;
    int   j0   = jb * 32 + lm, j1 = j0 + 16;
    float sqj0 = sq2[b * L + j0];
    float sqj1 = sq2[b * L + j1];
    #pragma unroll
    for (int r = 0; r < 8; ++r) {
        int   i0 = ib * 32 + hi * 8 + r;
        int   i1 = i0 + 16;
        float q0 = sq1[b * L + i0];
        float q1 = sq1[b * L + i1];
        Cb[(size_t)i0 * L + j0] = sqrtf(fmaxf(q0 + sqj0 - 2.0f * acc00[r], 0.0f));
        Cb[(size_t)i0 * L + j1] = sqrtf(fmaxf(q0 + sqj1 - 2.0f * acc01[r], 0.0f));
        Cb[(size_t)i1 * L + j0] = sqrtf(fmaxf(q1 + sqj0 - 2.0f * acc10[r], 0.0f));
        Cb[(size_t)i1 * L + j1] = sqrtf(fmaxf(q1 + sqj1 - 2.0f * acc11[r], 0.0f));
    }
}

// ---------------------------------------------------------------------------
// Kernel 3: anti-diagonal wavefront DTW. One 1024-thread workgroup per batch,
// three rotating diagonals (length 1025) in LDS, one barrier per diagonal,
// next-diagonal cost prefetched before the LDS reads so the global-load
// latency hides under the barrier + VALU of the current step.
// ---------------------------------------------------------------------------
__global__ void __launch_bounds__(1024)
dtw_wavefront_kernel(const float* __restrict__ C, float* __restrict__ out) {
    __shared__ float buf[3][1028];
    int b   = blockIdx.x;
    int tid = threadIdx.x;
    int i   = tid + 1;                 // 1..1024

    buf[0][i] = BIGF;                  // d0: all BIG except [0]=0
    buf[1][i] = BIGF;                  // d1: all BIG
    if (tid == 0) { buf[0][0] = 0.0f; buf[1][0] = BIGF; buf[2][0] = BIGF; }
    __syncthreads();

    // Cb[j] = C[b][i-1][j-1]
    const float* Cb = C + (size_t)b * L * L + (size_t)(i - 1) * L - 1;

    int p2 = 0, p1 = 1, cu = 2;
    int   j0    = 2 - i;
    float c_cur = (j0 >= 1 && j0 <= L) ? Cb[j0] : BIGF;

    for (int k = 2; k <= 2 * L; ++k) {
        int   jn     = k + 1 - i;      // prefetch next diagonal's cost
        float c_next = (k < 2 * L && jn >= 1 && jn <= L) ? Cb[jn] : BIGF;

        int  j     = k - i;
        bool valid = (j >= 1) && (j <= L);
        float m = fminf(buf[p2][i - 1], fminf(buf[p1][i - 1], buf[p1][i]));
        buf[cu][i] = valid ? (c_cur + m) : BIGF;
        if (tid == 0) buf[cu][0] = BIGF;
        __syncthreads();

        int t = p2; p2 = p1; p1 = cu; cu = t;
        c_cur = c_next;
    }

    // last diagonal (k = 2048) now sits in buf[p1]; D[L1,L2] = buf[p1][L].
    // softmax over a 1-element weight vector is exactly 1.0.
    if (tid == L - 1) {
        float d = buf[p1][L] / (float)(2 * L);
        out[b] = 1.0f / (1.0f + d);
    }
}

// ---------------------------------------------------------------------------
extern "C" void kernel_launch(void* const* d_in, const int* in_sizes, int n_in,
                              void* d_out, int out_size, void* d_ws, size_t ws_size,
                              hipStream_t stream) {
    const float* seq1 = (const float*)d_in[0];   // (32,1024,128) fp32
    const float* seq2 = (const float*)d_in[1];   // (32,1024,128) fp32
    // d_in[2] = scale_weights: softmax of 1 element == 1.0, not needed.
    float* out = (float*)d_out;                  // (32,) fp32

    // Workspace layout: sq1 | sq2 | h1(f16) | h2(f16) | C(fp32, 128 MiB)
    float*     sq1 = (float*)d_ws;                         // 32*1024
    float*     sq2 = sq1 + (size_t)BATCH * L;              // 32*1024
    _Float16*  h1  = (_Float16*)(sq2 + (size_t)BATCH * L); // 32*1024*128 f16
    _Float16*  h2  = h1 + (size_t)BATCH * L * F;           // 32*1024*128 f16
    float*     C   = (float*)(h2 + (size_t)BATCH * L * F); // 32*1024*1024 f32

    int  nrows = BATCH * L;                                // 32768
    dim3 nb((nrows + 7) / 8);                              // 8 waves / block
    dtw_prep_kernel<<<nb, 256, 0, stream>>>(seq1, sq1, h1, nrows);
    dtw_prep_kernel<<<nb, 256, 0, stream>>>(seq2, sq2, h2, nrows);

    dtw_cost_gemm_kernel<<<BATCH * 32 * 4, 256, 0, stream>>>(h1, h2, sq1, sq2, C);

    dtw_wavefront_kernel<<<BATCH, 1024, 0, stream>>>(C, out);
}
